// ExpertGroup_64089501991419
// MI455X (gfx1250) — matrix operand order repaired and checked by reading, not compiled, not run on hardware
//
#include <hip/hip_runtime.h>
#include <math.h>

// ---- problem constants (match reference) ----
#define BB 4
#define SS 2048
#define DD 1024
#define HH 2048
#define ADIM 128
#define EE 8
#define NTOK (BB*SS)   // 8192

typedef __attribute__((ext_vector_type(16))) __bf16 bf16x16;
typedef __attribute__((ext_vector_type(8)))  float  f32x8;
typedef int v4i_vs __attribute__((vector_size(16)));   // matches async-LDS builtin pointee

typedef struct __attribute__((aligned(16), may_alias)) { unsigned int x, y, z, w; } U4;

union Frag { U4 q[2]; bf16x16 v; };

#if __has_builtin(__builtin_amdgcn_global_load_async_to_lds_b128) && \
    __has_builtin(__builtin_amdgcn_s_wait_asynccnt)
#define HAVE_ASYNC_LDS 1
#else
#define HAVE_ASYNC_LDS 0
#endif

#define AS1 __attribute__((address_space(1)))
#define AS3 __attribute__((address_space(3)))

__device__ __forceinline__ unsigned short f2bf(float f) {
    unsigned int u = __float_as_uint(f);
    unsigned int r = u + 0x7FFFu + ((u >> 16) & 1u);   // round-to-nearest-even
    return (unsigned short)(r >> 16);
}

__device__ __forceinline__ f32x8 wmma_bf16(const Frag& a, const Frag& b, f32x8 c) {
    return __builtin_amdgcn_wmma_f32_16x16x32_bf16(false, a.v, false, b.v,
                                                   (short)0, c, false, false);
}

__device__ __forceinline__ float silu_f(float x) { return x / (1.f + __expf(-x)); }

// ---------------- epilogue modes ----------------
enum { EPI_BIAS = 0, EPI_SILU_MUL = 1, EPI_ADD_SCALED = 2, EPI_ROWSCALE = 3 };

// ---------------- tiled bf16 WMMA GEMM ----------------
// C[M,Nn] = epi( A[M,K](bf16) @ Bw[Nn,K]^T(bf16) )
// block tile 128x128, 8 waves (4 along M, 2 along N), wave tile 32x64.
// Double-buffered LDS pipeline fed by GLOBAL_LOAD_ASYNC_TO_LDS_B128 (ASYNCcnt).
template <int EPI, bool SHADOW>
__global__ __launch_bounds__(256)
void gemm_bf16(const unsigned short* __restrict__ A,
               const unsigned short* __restrict__ Bw,
               const float* __restrict__ bias,
               const float* __restrict__ other,
               const float* __restrict__ rowscale,
               float alpha,
               float* __restrict__ out,
               unsigned short* __restrict__ outb,
               int M, int Nn, int K)
{
#if HAVE_ASYNC_LDS
    __shared__ unsigned short lA[2][128 * 40];   // row stride 40 bf16 (80B, 16B-aligned)
    __shared__ unsigned short lB[2][128 * 40];
#else
    __shared__ unsigned short lA[1][128 * 40];
    __shared__ unsigned short lB[1][128 * 40];
#endif

    const int tid  = threadIdx.x;
    const int wave = tid >> 5;
    const int lane = tid & 31;
    const int wm = wave & 3;      // 0..3 -> M sub-tile
    const int wn = wave >> 2;     // 0..1 -> N sub-tile
    const int m0 = blockIdx.y * 128;
    const int n0 = blockIdx.x * 128;

    const int mlo    = lane & 15;
    const int half_a = (lane < 16) ? 0 : 8;
    const int koff_b = (lane < 16) ? 0 : 8;

    f32x8 acc[2][4];
#pragma unroll
    for (int i = 0; i < 2; ++i)
#pragma unroll
        for (int j = 0; j < 4; ++j) acc[i][j] = (f32x8){};

    const int nk = K >> 5;

#if HAVE_ASYNC_LDS
    // -------- async double-buffered pipeline --------
    // each stage = 4 async-to-LDS b128 instructions per wave (ASYNCcnt += 4)
    auto stage_async = [&](int buf, int k0) {
#pragma unroll
        for (int i = 0; i < 2; ++i) {
            int idx = tid + i * 256;
            int row = idx >> 2;
            int seg = idx & 3;
            __builtin_amdgcn_global_load_async_to_lds_b128(
                (AS1 v4i_vs*)&A[(size_t)(m0 + row) * K + k0 + seg * 8],
                (AS3 v4i_vs*)&lA[buf][row * 40 + seg * 8], 0, 0);
            __builtin_amdgcn_global_load_async_to_lds_b128(
                (AS1 v4i_vs*)&Bw[(size_t)(n0 + row) * K + k0 + seg * 8],
                (AS3 v4i_vs*)&lB[buf][row * 40 + seg * 8], 0, 0);
        }
    };
    stage_async(0, 0);
#endif

    for (int kt = 0; kt < nk; ++kt) {
        const int k0 = kt << 5;
#if HAVE_ASYNC_LDS
        const int cur = kt & 1;
        const bool has_next = (kt + 1 < nk);
        if (has_next) stage_async(cur ^ 1, k0 + 32);
        // current tile's 4 loads (issued last iteration / prologue) are the
        // oldest outstanding; leave the 4 just-issued in flight.
        if (has_next) __builtin_amdgcn_s_wait_asynccnt(4);
        else          __builtin_amdgcn_s_wait_asynccnt(0);
        __syncthreads();
#else
        const int cur = 0;
#pragma unroll
        for (int i = 0; i < 2; ++i) {
            int idx = tid + i * 256;
            int row = idx >> 2;
            int seg = idx & 3;
            *(U4*)&lA[0][row * 40 + seg * 8] =
                *(const U4*)&A[(size_t)(m0 + row) * K + k0 + seg * 8];
            *(U4*)&lB[0][row * 40 + seg * 8] =
                *(const U4*)&Bw[(size_t)(n0 + row) * K + k0 + seg * 8];
        }
        if (k0 + 32 < K) {  // prefetch next K tile -> global_prefetch_b8
            __builtin_prefetch(&A[(size_t)(m0 + (tid >> 1)) * K + k0 + 32], 0, 1);
            __builtin_prefetch(&Bw[(size_t)(n0 + (tid >> 1)) * K + k0 + 32], 0, 1);
        }
        __syncthreads();
#endif

        Frag fa[2], fb[4];
#pragma unroll
        for (int mt = 0; mt < 2; ++mt) {
            const unsigned short* p = &lA[cur][(wm * 32 + mt * 16 + mlo) * 40 + half_a];
            fa[mt].q[0] = *(const U4*)p;
            fa[mt].q[1] = *(const U4*)(p + 16);
        }
#pragma unroll
        for (int nt = 0; nt < 4; ++nt) {
            const unsigned short* p = &lB[cur][(wn * 64 + nt * 16 + mlo) * 40 + koff_b];
            fb[nt].q[0] = *(const U4*)p;
            fb[nt].q[1] = *(const U4*)(p + 16);
        }
#pragma unroll
        for (int mt = 0; mt < 2; ++mt)
#pragma unroll
            for (int nt = 0; nt < 4; ++nt)
                acc[mt][nt] = wmma_bf16(fa[mt], fb[nt], acc[mt][nt]);
        __syncthreads();
    }

    // epilogue
#pragma unroll
    for (int mt = 0; mt < 2; ++mt)
#pragma unroll
        for (int nt = 0; nt < 4; ++nt)
#pragma unroll
            for (int r = 0; r < 8; ++r) {
                int row = m0 + wm * 32 + mt * 16 + r + ((lane < 16) ? 0 : 8);
                int col = n0 + wn * 64 + nt * 16 + mlo;
                size_t idx = (size_t)row * Nn + col;
                float v = acc[mt][nt][r];
                if (EPI == EPI_BIAS) {
                    if (bias) v += bias[col];
                } else if (EPI == EPI_SILU_MUL) {
                    float t = v + (bias ? bias[col] : 0.f);
                    v = silu_f(t) * other[idx];
                } else if (EPI == EPI_ADD_SCALED) {
                    v = other[idx] + alpha * v;
                } else if (EPI == EPI_ROWSCALE) {
                    v = (v + (bias ? bias[col] : 0.f)) * rowscale[row];
                }
                out[idx] = v;
                if (SHADOW) outb[idx] = f2bf(v);
            }
}

// ---------------- adapter attention (per batch) ----------------
// adapt[b,s,:] = sum_t silu(clip(ain[b,s]·aout[b,t], -5, 5)) * ain[b,t,:]
__global__ __launch_bounds__(256)
void attn_kernel(const unsigned short* __restrict__ ain_b,
                 const unsigned short* __restrict__ aout_b,
                 float* __restrict__ adapt,
                 unsigned short* __restrict__ adapt_bf)
{
    __shared__ unsigned short lK[32 * 136];   // K tile [t=32][a=128], stride 136
    __shared__ unsigned short lVT[128 * 40];  // V^T tile [a=128][t=32], stride 40
    __shared__ unsigned short lS[8 * 16 * 40];// per-wave score scratch [m=16][t=32]

    const int tid  = threadIdx.x;
    const int wave = tid >> 5;
    const int lane = tid & 31;
    const int mlo    = lane & 15;
    const int half_a = (lane < 16) ? 0 : 8;
    const int koff_b = (lane < 16) ? 0 : 8;

    const int batch = blockIdx.y;
    const int mrow  = blockIdx.x * 128 + wave * 16;  // wave's s-row base within batch
    const size_t tok0 = (size_t)batch * SS;

    // Q fragments: adapt_in rows [mrow..mrow+16), K = 128 split into 4 x 32
    Frag fq[4];
#pragma unroll
    for (int ka = 0; ka < 4; ++ka) {
        const unsigned short* p = &ain_b[(tok0 + mrow + mlo) * ADIM + ka * 32 + half_a];
        fq[ka].q[0] = *(const U4*)p;
        fq[ka].q[1] = *(const U4*)(p + 16);
    }

    f32x8 acc[8];
#pragma unroll
    for (int i = 0; i < 8; ++i) acc[i] = (f32x8){};

    for (int t0 = 0; t0 < SS; t0 += 32) {
        // stage: K tile as-is, V tile transposed
#pragma unroll
        for (int i = 0; i < 2; ++i) {
            int idx = tid + i * 256;
            int row = idx >> 4;   // t_local 0..31
            int seg = idx & 15;   // 16 segments of 8 along a
            U4 qk = *(const U4*)&aout_b[(tok0 + t0 + row) * ADIM + seg * 8];
            *(U4*)&lK[row * 136 + seg * 8] = qk;
            U4 qv = *(const U4*)&ain_b[(tok0 + t0 + row) * ADIM + seg * 8];
            unsigned int vv[4] = { qv.x, qv.y, qv.z, qv.w };
#pragma unroll
            for (int j = 0; j < 4; ++j) {
                lVT[(seg * 8 + j * 2)     * 40 + row] = (unsigned short)(vv[j] & 0xffffu);
                lVT[(seg * 8 + j * 2 + 1) * 40 + row] = (unsigned short)(vv[j] >> 16);
            }
        }
        __syncthreads();

        // scores: [16 s] x [32 t], K = a = 128
        f32x8 sc[2];
        sc[0] = (f32x8){}; sc[1] = (f32x8){};
#pragma unroll
        for (int nt = 0; nt < 2; ++nt)
#pragma unroll
            for (int ka = 0; ka < 4; ++ka) {
                Frag fb;
                const unsigned short* p = &lK[(nt * 16 + mlo) * 136 + ka * 32 + koff_b];
                fb.q[0] = *(const U4*)p;
                fb.q[1] = *(const U4*)(p + 16);
                sc[nt] = wmma_bf16(fq[ka], fb, sc[nt]);
            }

        // silu(clip) and re-layout C->A via per-wave LDS scratch
        unsigned short* myS = &lS[wave * 16 * 40];
#pragma unroll
        for (int nt = 0; nt < 2; ++nt)
#pragma unroll
            for (int r = 0; r < 8; ++r) {
                float x = sc[nt][r];
                x = fminf(fmaxf(x, -5.f), 5.f);
                x = silu_f(x);
                int m = r + ((lane < 16) ? 0 : 8);
                myS[m * 40 + nt * 16 + mlo] = f2bf(x);
            }
        __syncthreads();

        // adapt accumulate: scoresA(16x32) @ V^T-B(32x128)
        Frag fs;
        {
            const unsigned short* p = &myS[mlo * 40 + half_a];
            fs.q[0] = *(const U4*)p;
            fs.q[1] = *(const U4*)(p + 16);
        }
#pragma unroll
        for (int nt = 0; nt < 8; ++nt) {
            Frag fb;
            const unsigned short* p = &lVT[(nt * 16 + mlo) * 40 + koff_b];
            fb.q[0] = *(const U4*)p;
            fb.q[1] = *(const U4*)(p + 16);
            acc[nt] = wmma_bf16(fs, fb, acc[nt]);
        }
        __syncthreads();
    }

#pragma unroll
    for (int nt = 0; nt < 8; ++nt)
#pragma unroll
        for (int r = 0; r < 8; ++r) {
            int row = mrow + r + ((lane < 16) ? 0 : 8);
            int col = nt * 16 + mlo;
            size_t idx = (tok0 + row) * ADIM + col;
            float v = acc[nt][r];
            adapt[idx]    = v;
            adapt_bf[idx] = f2bf(v);
        }
}

// ---------------- LayerNorm over 128 cols, one row per block ----------------
__global__ __launch_bounds__(128)
void ln128_kernel(const float* __restrict__ in, const float* __restrict__ g,
                  const float* __restrict__ bia, float* __restrict__ out,
                  unsigned short* __restrict__ outb)
{
    __shared__ float red[128];
    const int row = blockIdx.x, tid = threadIdx.x;
    const size_t base = (size_t)row * ADIM;
    float v = in[base + tid];
    red[tid] = v; __syncthreads();
    for (int s = 64; s; s >>= 1) { if (tid < s) red[tid] += red[tid + s]; __syncthreads(); }
    float mean = red[0] * (1.f / 128.f); __syncthreads();
    float d = v - mean;
    red[tid] = d * d; __syncthreads();
    for (int s = 64; s; s >>= 1) { if (tid < s) red[tid] += red[tid + s]; __syncthreads(); }
    float var = red[0] * (1.f / 128.f);
    float y = d * rsqrtf(var + 1e-5f) * g[tid] + bia[tid];
    out[base + tid]  = y;
    outb[base + tid] = f2bf(y);
}

// ---------------- per-expert LN + weighted combine ----------------
__global__ __launch_bounds__(128)
void expert_combine_kernel(const float* __restrict__ hbuf,   // [E, NTOK, AD]
                           const float* __restrict__ ew,     // [NTOK, E]
                           const float* __restrict__ eg,     // [E, AD]
                           const float* __restrict__ eb,     // [E, AD]
                           float* __restrict__ hw,
                           unsigned short* __restrict__ hwb)
{
    __shared__ float red[128];
    const int n = blockIdx.x, tid = threadIdx.x;
    float accv = 0.f;
    for (int e = 0; e < EE; ++e) {
        size_t base = ((size_t)e * NTOK + n) * ADIM;
        float v = hbuf[base + tid];
        red[tid] = v; __syncthreads();
        for (int s = 64; s; s >>= 1) { if (tid < s) red[tid] += red[tid + s]; __syncthreads(); }
        float mean = red[0] * (1.f / 128.f); __syncthreads();
        float d = v - mean;
        red[tid] = d * d; __syncthreads();
        for (int s = 64; s; s >>= 1) { if (tid < s) red[tid] += red[tid + s]; __syncthreads(); }
        float var = red[0] * (1.f / 128.f); __syncthreads();
        float y = d * rsqrtf(var + 1e-5f) * eg[e * ADIM + tid] + eb[e * ADIM + tid];
        float w = ew[(size_t)n * EE + e];
        w = (w > 0.f) ? w : 0.f;
        accv += w * y;
    }
    size_t o = (size_t)n * ADIM + tid;
    hw[o]  = accv;
    hwb[o] = f2bf(accv);
}

// ---------------- small utility kernels ----------------
__global__ void conv_bf16_kernel(const float* __restrict__ in,
                                 unsigned short* __restrict__ out, size_t n)
{
    size_t i = (size_t)blockIdx.x * 256 + threadIdx.x;
    if (i < n) out[i] = f2bf(in[i]);
}

__global__ void transconv_kernel(const float* __restrict__ in,   // [R,C]
                                 unsigned short* __restrict__ out, // [C,R]
                                 int R, int C)
{
    size_t i = (size_t)blockIdx.x * 256 + threadIdx.x;
    if (i < (size_t)R * C) {
        int r = (int)(i / C), c = (int)(i % C);
        out[(size_t)c * R + r] = f2bf(in[i]);
    }
}

__global__ void rowsum_kernel(const float* __restrict__ ew, float* __restrict__ rs)
{
    int n = blockIdx.x * 256 + threadIdx.x;
    if (n < NTOK) {
        float s = 0.f;
        for (int e = 0; e < EE; ++e) s += ew[(size_t)n * EE + e];
        rs[n] = s;
    }
}

// ---------------- host orchestration ----------------
extern "C" void kernel_launch(void* const* d_in, const int* in_sizes, int n_in,
                              void* d_out, int out_size, void* d_ws, size_t ws_size,
                              hipStream_t stream)
{
    (void)in_sizes; (void)n_in; (void)out_size; (void)ws_size;
    const float* x     = (const float*)d_in[0];
    const float* ew    = (const float*)d_in[1];
    const float* Wu    = (const float*)d_in[2];
    const float* bu    = (const float*)d_in[3];
    const float* Wg    = (const float*)d_in[4];
    const float* bg    = (const float*)d_in[5];
    const float* Wd    = (const float*)d_in[6];
    const float* bd    = (const float*)d_in[7];
    const float* Wpre  = (const float*)d_in[8];
    const float* bpre  = (const float*)d_in[9];
    const float* Wpost = (const float*)d_in[10];
    const float* bpost = (const float*)d_in[11];
    const float* ln_g  = (const float*)d_in[12];
    const float* ln_b  = (const float*)d_in[13];
    const float* Wap   = (const float*)d_in[14];
    const float* Aexp  = (const float*)d_in[15];
    const float* eg    = (const float*)d_in[16];
    const float* eb    = (const float*)d_in[17];
    const float* Wp    = (const float*)d_in[18];
    const float* Wo    = (const float*)d_in[19];
    float* out = (float*)d_out;

    // bump allocator on workspace
    char* wsb = (char*)d_ws;
    size_t off = 0;
    auto alloc = [&](size_t bytes) -> void* {
        off = (off + 255) & ~(size_t)255;
        void* p = wsb + off;
        off += bytes;
        return p;
    };
    typedef unsigned short ush;
    ush* xb      = (ush*)alloc((size_t)NTOK * DD * 2);
    ush* Wu_b    = (ush*)alloc((size_t)HH * DD * 2);
    ush* Wg_b    = (ush*)alloc((size_t)HH * DD * 2);
    ush* Wd_b    = (ush*)alloc((size_t)DD * HH * 2);
    ush* Wpre_b  = (ush*)alloc((size_t)ADIM * DD * 2);
    ush* Wpost_b = (ush*)alloc((size_t)ADIM * HH * 2);
    ush* Wap_b   = (ush*)alloc((size_t)HH * ADIM * 2);
    ush* A_b     = (ush*)alloc((size_t)EE * ADIM * ADIM * 2);
    ush* Wo_b    = (ush*)alloc((size_t)DD * HH * 2);
    ush* WpT_b   = (ush*)alloc((size_t)ADIM * HH * 2);
    ush* hid_b   = (ush*)alloc((size_t)NTOK * HH * 2);
    ush* pre_b   = (ush*)alloc((size_t)NTOK * ADIM * 2);
    ush* ain_b   = (ush*)alloc((size_t)NTOK * ADIM * 2);
    ush* aout_b  = (ush*)alloc((size_t)NTOK * ADIM * 2);
    ush* adapt_b = (ush*)alloc((size_t)NTOK * ADIM * 2);
    ush* hw_b    = (ush*)alloc((size_t)NTOK * ADIM * 2);
    ush* Wc_b    = (ush*)alloc((size_t)DD * ADIM * 2);
    float* up    = (float*)alloc((size_t)NTOK * HH * 4);
    float* hid   = (float*)alloc((size_t)NTOK * HH * 4);
    float* pre   = (float*)alloc((size_t)NTOK * ADIM * 4);
    float* ain   = (float*)alloc((size_t)NTOK * ADIM * 4);
    float* araw  = (float*)alloc((size_t)NTOK * ADIM * 4);
    float* aoutf = (float*)alloc((size_t)NTOK * ADIM * 4);
    float* adapt = (float*)alloc((size_t)NTOK * ADIM * 4);
    float* hbuf  = (float*)alloc((size_t)EE * NTOK * ADIM * 4);
    float* hw    = (float*)alloc((size_t)NTOK * ADIM * 4);
    float* Wc    = (float*)alloc((size_t)DD * ADIM * 4);
    float* rsum  = (float*)alloc((size_t)NTOK * 4);

    auto cgrid = [](size_t n) { return dim3((unsigned)((n + 255) / 256)); };
    // f32 -> bf16 conversions
    conv_bf16_kernel<<<cgrid((size_t)NTOK * DD), 256, 0, stream>>>(x, xb, (size_t)NTOK * DD);
    conv_bf16_kernel<<<cgrid((size_t)HH * DD), 256, 0, stream>>>(Wu, Wu_b, (size_t)HH * DD);
    conv_bf16_kernel<<<cgrid((size_t)HH * DD), 256, 0, stream>>>(Wg, Wg_b, (size_t)HH * DD);
    conv_bf16_kernel<<<cgrid((size_t)DD * HH), 256, 0, stream>>>(Wd, Wd_b, (size_t)DD * HH);
    conv_bf16_kernel<<<cgrid((size_t)ADIM * DD), 256, 0, stream>>>(Wpre, Wpre_b, (size_t)ADIM * DD);
    conv_bf16_kernel<<<cgrid((size_t)ADIM * HH), 256, 0, stream>>>(Wpost, Wpost_b, (size_t)ADIM * HH);
    conv_bf16_kernel<<<cgrid((size_t)HH * ADIM), 256, 0, stream>>>(Wap, Wap_b, (size_t)HH * ADIM);
    conv_bf16_kernel<<<cgrid((size_t)EE * ADIM * ADIM), 256, 0, stream>>>(Aexp, A_b, (size_t)EE * ADIM * ADIM);
    conv_bf16_kernel<<<cgrid((size_t)DD * HH), 256, 0, stream>>>(Wo, Wo_b, (size_t)DD * HH);
    transconv_kernel<<<cgrid((size_t)HH * ADIM), 256, 0, stream>>>(Wp, WpT_b, HH, ADIM);
    rowsum_kernel<<<cgrid(NTOK), 256, 0, stream>>>(ew, rsum);

    // up = x@Wu^T + bu
    gemm_bf16<EPI_BIAS, false><<<dim3(HH / 128, NTOK / 128), 256, 0, stream>>>(
        xb, Wu_b, bu, nullptr, nullptr, 1.f, up, nullptr, NTOK, HH, DD);
    // hidden = silu(x@Wg^T + bg) * up   (+ bf16 shadow)
    gemm_bf16<EPI_SILU_MUL, true><<<dim3(HH / 128, NTOK / 128), 256, 0, stream>>>(
        xb, Wg_b, bg, up, nullptr, 1.f, hid, hid_b, NTOK, HH, DD);
    // pre = x@Wpre^T + bpre  (+ bf16 shadow, reused by expert branch)
    gemm_bf16<EPI_BIAS, true><<<dim3(ADIM / 128, NTOK / 128), 256, 0, stream>>>(
        xb, Wpre_b, bpre, nullptr, nullptr, 1.f, pre, pre_b, NTOK, ADIM, DD);
    // adapt_in = LN(pre)
    ln128_kernel<<<NTOK, 128, 0, stream>>>(pre, ln_g, ln_b, ain, ain_b);
    // adapt_out = LN(hidden@Wpost^T + bpost)
    gemm_bf16<EPI_BIAS, false><<<dim3(ADIM / 128, NTOK / 128), 256, 0, stream>>>(
        hid_b, Wpost_b, bpost, nullptr, nullptr, 1.f, araw, nullptr, NTOK, ADIM, HH);
    ln128_kernel<<<NTOK, 128, 0, stream>>>(araw, ln_g, ln_b, aoutf, aout_b);
    // adapter attention
    attn_kernel<<<dim3(SS / 128, BB), 256, 0, stream>>>(ain_b, aout_b, adapt, adapt_b);
    // hidden += 0.1 * adapt@Wap^T  (in-place, refresh bf16 shadow)
    gemm_bf16<EPI_ADD_SCALED, true><<<dim3(HH / 128, NTOK / 128), 256, 0, stream>>>(
        adapt_b, Wap_b, nullptr, hid, nullptr, 0.1f, hid, hid_b, NTOK, HH, ADIM);
    // out = (hidden@Wd^T + bd) * rowsum   (shared * sum(w))
    gemm_bf16<EPI_ROWSCALE, false><<<dim3(DD / 128, NTOK / 128), 256, 0, stream>>>(
        hid_b, Wd_b, bd, nullptr, rsum, 1.f, out, nullptr, NTOK, DD, HH);
    // expert adapters: h_e = pre@A_e^T
    for (int e = 0; e < EE; ++e)
        gemm_bf16<EPI_BIAS, false><<<dim3(ADIM / 128, NTOK / 128), 256, 0, stream>>>(
            pre_b, A_b + (size_t)e * ADIM * ADIM, nullptr, nullptr, nullptr, 1.f,
            hbuf + (size_t)e * NTOK * ADIM, nullptr, NTOK, ADIM, ADIM);
    // hw = sum_e relu(w_e) * LN_e(h_e)
    expert_combine_kernel<<<NTOK, 128, 0, stream>>>(hbuf, ew, eg, eb, hw, hw_b);
    // Wc = Wo @ Wp  (B operand from transposed Wp)
    gemm_bf16<EPI_BIAS, true><<<dim3(ADIM / 128, DD / 128), 256, 0, stream>>>(
        Wo_b, WpT_b, nullptr, nullptr, nullptr, 1.f, Wc, Wc_b, DD, ADIM, HH);
    // out += 0.1 * hw@Wc^T
    gemm_bf16<EPI_ADD_SCALED, false><<<dim3(DD / 128, NTOK / 128), 256, 0, stream>>>(
        hw_b, Wc_b, nullptr, out, nullptr, 0.1f, out, nullptr, NTOK, DD, ADIM);
}
